// DirSageConv_25829933318545
// MI455X (gfx1250) — compile-verified, matrix operand order checked by reading, not run on hardware
//
#include <hip/hip_runtime.h>
#include <stdint.h>

#define NNODES 100000
#define NEDGES 640000
#define DIM    128

typedef __bf16 v16bf __attribute__((ext_vector_type(16)));
typedef float  v8f   __attribute__((ext_vector_type(8)));

// ---------------- workspace layout (see kernel_launch) ----------------
// sum_st : N*128 f32      (scatter-sum, src->dst)
// sum_ts : N*128 f32      (scatter-sum, dst->src)
// cnt_st : N f32
// cnt_ts : N f32
// b_tot  : 128 f32        (b_lin + 0.5*(b_st+b_ts))
// wfrag  : 12*8*32*16 bf16 (combined K=384 weight, WMMA B-fragment layout)

#define WFRAG_ELEMS (12 * 8 * 32 * 16)           // 49152 bf16 = 96 KB

__global__ void zero_ws_kernel(float* __restrict__ ws, long long n) {
  long long i = (long long)blockIdx.x * blockDim.x + threadIdx.x;
  long long stride = (long long)gridDim.x * blockDim.x;
  for (; i < n; i += stride) ws[i] = 0.0f;
}

__global__ void prep_kernel(const float* __restrict__ Wl, const float* __restrict__ bl,
                            const float* __restrict__ Ws, const float* __restrict__ bs,
                            const float* __restrict__ Wt, const float* __restrict__ bt,
                            float* __restrict__ b_tot, uint16_t* __restrict__ wfrag) {
  int tid = blockIdx.x * blockDim.x + threadIdx.x;
  if (tid < DIM) b_tot[tid] = bl[tid] + 0.5f * (bs[tid] + bt[tid]);
  if (tid >= 12 * 8 * 32) return;
  int fid   = tid >> 5;          // fragment id = ktile*8 + ntile
  int lane  = tid & 31;
  int ktile = fid >> 3;
  int ntile = fid & 7;
  int n     = ntile * 16 + (lane & 15);
  int kbase = ktile * 32 + ((lane >> 4) * 8);  // lanes 16-31 hold K+8 half
  uint16_t* outp = wfrag + (long long)tid * 16;
#pragma unroll
  for (int s = 0; s < 16; ++s) {
    int k = kbase + (s < 8 ? s : s + 8);       // slots 8..15 are K+16..K+23
    float v;
    if (k < 128)       v =        Wl[k * DIM + n];
    else if (k < 256)  v = 0.5f * Ws[(k - 128) * DIM + n];
    else               v = 0.5f * Wt[(k - 256) * DIM + n];
    __bf16 h = (__bf16)v;
    outp[s] = __builtin_bit_cast(uint16_t, h);
  }
}

// one wave32 per edge; 4 floats per lane per direction
__global__ __launch_bounds__(256) void scatter_kernel(
    const float* __restrict__ x, const long long* __restrict__ ei,
    float* __restrict__ sum_st, float* __restrict__ sum_ts,
    float* __restrict__ cnt_st, float* __restrict__ cnt_ts) {
  int gtid = blockIdx.x * blockDim.x + threadIdx.x;
  int e    = gtid >> 5;
  int lane = gtid & 31;
  if (e >= NEDGES) return;
  int s = (int)ei[e];
  int d = (int)ei[NEDGES + e];
  const float4 xs = *(const float4*)(x + (long long)s * DIM + lane * 4);
  const float4 xd = *(const float4*)(x + (long long)d * DIM + lane * 4);
  float* ps = sum_st + (long long)d * DIM + lane * 4;
  float* pt = sum_ts + (long long)s * DIM + lane * 4;
  __hip_atomic_fetch_add(ps + 0, xs.x, __ATOMIC_RELAXED, __HIP_MEMORY_SCOPE_AGENT);
  __hip_atomic_fetch_add(ps + 1, xs.y, __ATOMIC_RELAXED, __HIP_MEMORY_SCOPE_AGENT);
  __hip_atomic_fetch_add(ps + 2, xs.z, __ATOMIC_RELAXED, __HIP_MEMORY_SCOPE_AGENT);
  __hip_atomic_fetch_add(ps + 3, xs.w, __ATOMIC_RELAXED, __HIP_MEMORY_SCOPE_AGENT);
  __hip_atomic_fetch_add(pt + 0, xd.x, __ATOMIC_RELAXED, __HIP_MEMORY_SCOPE_AGENT);
  __hip_atomic_fetch_add(pt + 1, xd.y, __ATOMIC_RELAXED, __HIP_MEMORY_SCOPE_AGENT);
  __hip_atomic_fetch_add(pt + 2, xd.z, __ATOMIC_RELAXED, __HIP_MEMORY_SCOPE_AGENT);
  __hip_atomic_fetch_add(pt + 3, xd.w, __ATOMIC_RELAXED, __HIP_MEMORY_SCOPE_AGENT);
  if (lane == 0) {
    __hip_atomic_fetch_add(cnt_st + d, 1.0f, __ATOMIC_RELAXED, __HIP_MEMORY_SCOPE_AGENT);
    __hip_atomic_fetch_add(cnt_ts + s, 1.0f, __ATOMIC_RELAXED, __HIP_MEMORY_SCOPE_AGENT);
  }
}

// fused triple-GEMM: out = [x | sum_st/deg | sum_ts/deg] (K=384) @ W' + b'
// weights staged once per block in LDS (96 KB of the 320 KB WGP budget);
// one wave32 -> 16 nodes x 128 outputs; 12 ktiles x 8 ntiles WMMA bf16
__global__ __launch_bounds__(256) void node_gemm_kernel(
    const float* __restrict__ x,
    const float* __restrict__ sum_st, const float* __restrict__ sum_ts,
    const float* __restrict__ cnt_st, const float* __restrict__ cnt_ts,
    const float* __restrict__ b_tot,  const uint16_t* __restrict__ wfrag,
    float* __restrict__ out) {
  extern __shared__ uint16_t smem[];   // WFRAG_ELEMS bf16

  // cooperative weight stage: 6144 x 16B, 24 per thread
  {
    const uint4* gsrc = (const uint4*)wfrag;
    uint4* ldst = (uint4*)smem;
#pragma unroll
    for (int i = 0; i < WFRAG_ELEMS / 8 / 256; ++i)
      ldst[threadIdx.x + i * 256] = gsrc[threadIdx.x + i * 256];
  }
  __syncthreads();

  int wave = threadIdx.x >> 5;
  int lane = threadIdx.x & 31;
  int nodebase = (blockIdx.x * 8 + wave) * 16;
  if (nodebase >= NNODES) return;          // wave-uniform: EXEC stays all-ones

  int m    = lane & 15;                    // A: lanes 0-15 and 16-31 both hold M=0..15
  int row  = nodebase + m;
  int rowc = row < NNODES ? row : NNODES - 1;
  int khalf = (lane >> 4) * 8;             // lanes 16-31 take the K+8 half

  float cst = cnt_st[rowc]; cst = cst > 1.0f ? cst : 1.0f;
  float cts = cnt_ts[rowc]; cts = cts > 1.0f ? cts : 1.0f;
  float inv_st = 1.0f / cst;
  float inv_ts = 1.0f / cts;

  v8f acc[8];
  v8f zero = {0.f, 0.f, 0.f, 0.f, 0.f, 0.f, 0.f, 0.f};
#pragma unroll
  for (int i = 0; i < 8; ++i) acc[i] = zero;

#pragma unroll
  for (int ktile = 0; ktile < 12; ++ktile) {
    const float* src;
    float scale;
    int koff = (ktile & 3) * 32 + khalf;
    if (ktile < 4)      { src = x      + (long long)rowc * DIM + koff; scale = 1.0f;   }
    else if (ktile < 8) { src = sum_st + (long long)rowc * DIM + koff; scale = inv_st; }
    else                { src = sum_ts + (long long)rowc * DIM + koff; scale = inv_ts; }

    float vals[16];
    *(float4*)(vals + 0)  = *(const float4*)(src + 0);
    *(float4*)(vals + 4)  = *(const float4*)(src + 4);
    *(float4*)(vals + 8)  = *(const float4*)(src + 16);   // slots 8..15 = K+16..K+23
    *(float4*)(vals + 12) = *(const float4*)(src + 20);
    v16bf a;
#pragma unroll
    for (int i = 0; i < 16; ++i) a[i] = (__bf16)(vals[i] * scale);

    const uint16_t* wbase = smem + (ktile * 8 * 32 + lane) * 16;
#pragma unroll
    for (int ntile = 0; ntile < 8; ++ntile) {
      v16bf b = *(const v16bf*)(wbase + ntile * 32 * 16);
      acc[ntile] = __builtin_amdgcn_wmma_f32_16x16x32_bf16(
          false, a, false, b, (short)0, acc[ntile], false, false);
    }
  }

  // C/D layout: lanes 0-15 -> N=lane, M=r ; lanes 16-31 -> N=lane-16, M=8+r
  int mrow0 = (lane >> 4) * 8;
#pragma unroll
  for (int ntile = 0; ntile < 8; ++ntile) {
    int col = ntile * 16 + m;
    float bias = b_tot[col];
#pragma unroll
    for (int r = 0; r < 8; ++r) {
      int node = nodebase + mrow0 + r;
      if (node < NNODES) out[(long long)node * DIM + col] = acc[ntile][r] + bias;
    }
  }
}

extern "C" void kernel_launch(void* const* d_in, const int* in_sizes, int n_in,
                              void* d_out, int out_size, void* d_ws, size_t ws_size,
                              hipStream_t stream) {
  const float*     x  = (const float*)d_in[0];
  const long long* ei = (const long long*)d_in[1];   // int64 edge_index, shape (2,E)
  const float* Wl = (const float*)d_in[2];
  const float* bl = (const float*)d_in[3];
  const float* Ws = (const float*)d_in[4];
  const float* bs = (const float*)d_in[5];
  const float* Wt = (const float*)d_in[6];
  const float* bt = (const float*)d_in[7];
  float* out = (float*)d_out;

  float* ws      = (float*)d_ws;
  float* sum_st  = ws;
  float* sum_ts  = ws + (long long)NNODES * DIM;
  float* cnt_st  = ws + 2LL * NNODES * DIM;
  float* cnt_ts  = cnt_st + NNODES;
  float* b_tot   = cnt_ts + NNODES;
  uint16_t* wfrag = (uint16_t*)(b_tot + DIM);

  long long nzero = 2LL * NNODES * DIM + 2LL * NNODES;
  zero_ws_kernel<<<2048, 256, 0, stream>>>(ws, nzero);

  prep_kernel<<<(12 * 8 * 32 + 255) / 256, 256, 0, stream>>>(
      Wl, bl, Ws, bs, Wt, bt, b_tot, wfrag);

  // one wave per edge: E*32 threads
  scatter_kernel<<<(NEDGES * 32) / 256, 256, 0, stream>>>(
      x, ei, sum_st, sum_ts, cnt_st, cnt_ts);

  node_gemm_kernel<<<(NNODES + 127) / 128, 256, WFRAG_ELEMS * sizeof(uint16_t), stream>>>(
      x, sum_st, sum_ts, cnt_st, cnt_ts, b_tot, wfrag, out);
}
